// Model_10660108829167
// MI455X (gfx1250) — matrix operand order repaired and checked by reading, not compiled, not run on hardware
//
#include <hip/hip_runtime.h>
#include <cstdint>
#include <cstddef>

// ---- model dims ----
#define Bq   2
#define Sq   2048
#define Cq   1024
#define Hq   16
#define HDq  64
#define Vq   32000

typedef __bf16 bf16_t;
typedef __attribute__((ext_vector_type(16))) __bf16 v16bf;
typedef __attribute__((ext_vector_type(8)))  float  v8f;

union BF16x16 { v16bf v; unsigned u[8]; unsigned short s[16]; };
union U16x8   { int4 v; unsigned short s[8]; };

__device__ __forceinline__ bf16_t tobf(float f) { return (bf16_t)f; }

// async global->LDS copy of 16 bytes per lane (tracked by ASYNCcnt)
__device__ __forceinline__ void async_cp_b128(const void* lds, const void* gptr)
{
    unsigned            la = (unsigned)(size_t)lds;            // LDS byte address
    unsigned long long  ga = (unsigned long long)(size_t)gptr; // global address
    asm volatile("global_load_async_to_lds_b128 %0, %1, off"
                 :: "v"(la), "v"(ga) : "memory");
}

// ---------------------------------------------------------------------------
// x = tok_emb[idx] + pos_emb  ->  bf16
// ---------------------------------------------------------------------------
__global__ __launch_bounds__(256) void embed_kernel(const int* __restrict__ idx,
                                                    const float* __restrict__ tok,
                                                    const float* __restrict__ pos,
                                                    bf16_t* __restrict__ x)
{
    size_t i = (size_t)blockIdx.x * 256 + threadIdx.x;
    if (i >= (size_t)Bq * Sq * Cq) return;
    int c = (int)(i % Cq);
    int s = (int)((i / Cq) % Sq);
    int b = (int)(i / ((size_t)Cq * Sq));
    int t = idx[b * Sq + s];
    x[i] = tobf(tok[(size_t)t * Cq + c] + pos[(size_t)s * Cq + c]);
}

// ---------------------------------------------------------------------------
// generic f32 -> bf16
// ---------------------------------------------------------------------------
__global__ __launch_bounds__(256) void cvt_kernel(const float* __restrict__ in,
                                                  bf16_t* __restrict__ out, long n)
{
    size_t i = (size_t)blockIdx.x * 256 + threadIdx.x;
    if (i < (size_t)n) out[i] = tobf(in[i]);
}

// ---------------------------------------------------------------------------
// Wq/Wk/Wv : (H, C, HD) f32  ->  [C][H*HD] bf16 (GEMM-ready column layout)
// ---------------------------------------------------------------------------
__global__ __launch_bounds__(256) void repack_qkv_kernel(const float* __restrict__ w,
                                                         bf16_t* __restrict__ out)
{
    size_t i = (size_t)blockIdx.x * 256 + threadIdx.x;
    if (i >= (size_t)Hq * Cq * HDq) return;
    int d = (int)(i % HDq);
    int c = (int)((i / HDq) % Cq);
    int h = (int)(i / ((size_t)HDq * Cq));
    out[(size_t)c * (Hq * HDq) + h * HDq + d] = tobf(w[i]);
}

// ---------------------------------------------------------------------------
// bf16 WMMA GEMM:  C[M,N] = A[M,K] * B[K,N]  (+bias, +relu), fp32 accum
// BM=BN=128, BK=32, 256 threads = 8 waves (wave32), wave tile 64x32.
// Double-buffered LDS; A staged via GLOBAL_LOAD_ASYNC_TO_LDS_B128 (ASYNCcnt),
// B staged via b128 register loads + transposed LDS scatter (pipelined).
// EPI: 0 = none, 1 = bias, 2 = bias+relu.  OBF: bf16 vs f32 output.
// ---------------------------------------------------------------------------
template<int EPI, bool OBF>
__global__ __launch_bounds__(256) void gemm_kernel(const bf16_t* __restrict__ A,
                                                   const bf16_t* __restrict__ Bm,
                                                   const float* __restrict__ bias,
                                                   void* __restrict__ Cp,
                                                   int M, int N, int K)
{
    __shared__ unsigned short As[2][128][32];   // A tile, row-major [m][k]
    __shared__ unsigned short Bs[2][128][32];   // B tile, transposed [n][k]

    const int tid  = threadIdx.x;
    const int lane = tid & 31;
    const int wave = tid >> 5;
    const int wm = wave & 1;                    // 2 waves along M
    const int wn = wave >> 1;                   // 4 waves along N
    const int lane_lo = lane & 15;
    const int half    = lane >> 4;
    const int bm = blockIdx.y * 128;
    const int bn = blockIdx.x * 128;

    v8f acc[4][2] = {};

    const int ar  = tid >> 1, ak  = (tid & 1) * 16;   // A staging: 32B/thread
    const int bkr = tid >> 3, bno = (tid & 7) * 16;   // B staging: 32B/thread

    const bf16_t* aptr = A  + (size_t)(bm + ar) * K + ak;
    const bf16_t* bptr = Bm + (size_t)bkr * N + bn + bno;

    const int niter = K >> 5;

    // ---- prologue: stage 0 ----
    async_cp_b128(&As[0][ar][ak],     aptr);
    async_cp_b128(&As[0][ar][ak + 8], aptr + 8);
    U16x8 brg0, brg1;
    brg0.v = *(const int4*)bptr;
    brg1.v = *(const int4*)(bptr + 8);
#pragma unroll
    for (int i = 0; i < 8; ++i) Bs[0][bno + i][bkr]     = brg0.s[i];
#pragma unroll
    for (int i = 0; i < 8; ++i) Bs[0][bno + 8 + i][bkr] = brg1.s[i];

    for (int it = 0; it < niter; ++it) {
        const int p = it & 1, q = p ^ 1;
        const bool more = (it + 1) < niter;

        // ---- issue next stage (async A copy + B register loads) ----
        if (more) {
            const bf16_t* an = aptr + (size_t)(it + 1) * 32;
            async_cp_b128(&As[q][ar][ak],     an);
            async_cp_b128(&As[q][ar][ak + 8], an + 8);
            const bf16_t* bnx = bptr + (size_t)(it + 1) * 32 * N;
            brg0.v = *(const int4*)bnx;
            brg1.v = *(const int4*)(bnx + 8);
        }

        // current stage's 2 async copies are the oldest outstanding ones
        if (more) asm volatile("s_wait_asynccnt 0x2" ::: "memory");
        else      asm volatile("s_wait_asynccnt 0x0" ::: "memory");
        __syncthreads();

        // ---- fragments from stage p ----
        BF16x16 af[4], bfr[2];
#pragma unroll
        for (int mi = 0; mi < 4; ++mi) {
            const int row = wm * 64 + mi * 16 + lane_lo;
#pragma unroll
            for (int v = 0; v < 8; ++v) {
                const int kb = (v >> 2) * 16 + half * 8 + (v & 3) * 2;   // A layout
                af[mi].u[v] = *(const unsigned*)&As[p][row][kb];
            }
        }
#pragma unroll
        for (int ni = 0; ni < 2; ++ni) {
            const int col = wn * 32 + ni * 16 + lane_lo;
#pragma unroll
            for (int v = 0; v < 8; ++v) {
                const int kb = (v >> 2) * 16 + half * 8 + (v & 3) * 2;   // B layout
                bfr[ni].u[v] = *(const unsigned*)&Bs[p][col][kb];
            }
        }
        // ---- 8 WMMAs per wave per K-step ----
#pragma unroll
        for (int mi = 0; mi < 4; ++mi)
#pragma unroll
            for (int ni = 0; ni < 2; ++ni)
                acc[mi][ni] = __builtin_amdgcn_wmma_f32_16x16x32_bf16(
                    false, af[mi].v, false, bfr[ni].v, (short)0, acc[mi][ni], false, false);

        // ---- publish next B tile into the free stage ----
        if (more) {
#pragma unroll
            for (int i = 0; i < 8; ++i) Bs[q][bno + i][bkr]     = brg0.s[i];
#pragma unroll
            for (int i = 0; i < 8; ++i) Bs[q][bno + 8 + i][bkr] = brg1.s[i];
        }
        __syncthreads();
    }

    // ---- epilogue ----
#pragma unroll
    for (int mi = 0; mi < 4; ++mi) {
#pragma unroll
        for (int ni = 0; ni < 2; ++ni) {
            const int col = bn + wn * 32 + ni * 16 + lane_lo;
            const float bv = (EPI >= 1) ? bias[col] : 0.f;
#pragma unroll
            for (int j = 0; j < 8; ++j) {
                const int row = bm + wm * 64 + mi * 16 + half * 8 + j;
                float x = acc[mi][ni][j] + bv;
                if (EPI == 2) x = fmaxf(x, 0.f);
                if (OBF) ((bf16_t*)Cp)[(size_t)row * N + col] = tobf(x);
                else     ((float*)Cp)[(size_t)row * N + col]  = x;
            }
        }
    }
}

// ---------------------------------------------------------------------------
// Flash-attention: one wave32 per (b, h, 16-query-row block).
// Q/K/V are bf16 in [b*S][H*HD] row-major (GEMM output layout).
// Online softmax in fp32, causal mask, P bounced through LDS to A-frag layout.
// ---------------------------------------------------------------------------
__global__ __launch_bounds__(32) void attn_kernel(const bf16_t* __restrict__ Q,
                                                  const bf16_t* __restrict__ K,
                                                  const bf16_t* __restrict__ V,
                                                  bf16_t* __restrict__ O)
{
    __shared__ bf16_t pls[16][32];

    const int lane    = threadIdx.x;
    const int lane_lo = lane & 15;
    const int half    = lane >> 4;
    const int blk  = blockIdx.x;
    const int qblk = blk % (Sq / 16);
    const int bh   = blk / (Sq / 16);
    const int h = bh % Hq, b = bh / Hq;
    const int q0 = qblk * 16;

    const bf16_t* Qb = Q + (size_t)b * Sq * Cq + (size_t)h * HDq;
    const bf16_t* Kb = K + (size_t)b * Sq * Cq + (size_t)h * HDq;
    const bf16_t* Vb = V + (size_t)b * Sq * Cq + (size_t)h * HDq;

    // Q fragments: A 16x32 layout, two K-steps cover HD=64
    BF16x16 aq[2];
#pragma unroll
    for (int ks = 0; ks < 2; ++ks)
#pragma unroll
        for (int v = 0; v < 8; ++v) {
            const int kb = ks * 32 + (v >> 2) * 16 + half * 8 + (v & 3) * 2;
            aq[ks].u[v] = *(const unsigned*)(Qb + (size_t)(q0 + lane_lo) * Cq + kb);
        }

    float mrow[8], lrow[8];
    v8f oacc[4] = {};
#pragma unroll
    for (int j = 0; j < 8; ++j) { mrow[j] = -1e30f; lrow[j] = 0.f; }

    const int nblk = (q0 + 15) / 32 + 1;     // 32-key blocks up to causal frontier
    for (int tb = 0; tb < nblk; ++tb) {
        const int t0 = tb * 32;

        // ---- scores: S(16x32) = Q(16x64) * K^T(64x32) ----
        v8f sacc[2] = {};
#pragma unroll
        for (int nt = 0; nt < 2; ++nt) {
            BF16x16 bk0, bk1;
            const int key = t0 + nt * 16 + lane_lo;
#pragma unroll
            for (int v = 0; v < 8; ++v) {
                const int kb = (v >> 2) * 16 + half * 8 + (v & 3) * 2;
                bk0.u[v] = *(const unsigned*)(Kb + (size_t)key * Cq + kb);
                bk1.u[v] = *(const unsigned*)(Kb + (size_t)key * Cq + 32 + kb);
            }
            sacc[nt] = __builtin_amdgcn_wmma_f32_16x16x32_bf16(
                false, aq[0].v, false, bk0.v, (short)0, sacc[nt], false, false);
            sacc[nt] = __builtin_amdgcn_wmma_f32_16x16x32_bf16(
                false, aq[1].v, false, bk1.v, (short)0, sacc[nt], false, false);
        }

        // ---- scale + causal mask + row max ----
        float rmax[8];
#pragma unroll
        for (int j = 0; j < 8; ++j) rmax[j] = -1e30f;
#pragma unroll
        for (int nt = 0; nt < 2; ++nt) {
            const int col = t0 + nt * 16 + lane_lo;
#pragma unroll
            for (int j = 0; j < 8; ++j) {
                const int row = q0 + half * 8 + j;
                float sv = sacc[nt][j] * 0.125f;       // HD^-0.5
                if (col > row) sv = -1e30f;
                sacc[nt][j] = sv;
                rmax[j] = fmaxf(rmax[j], sv);
            }
        }
#pragma unroll
        for (int j = 0; j < 8; ++j) {
            rmax[j] = fmaxf(rmax[j], __shfl_xor(rmax[j], 1, 32));
            rmax[j] = fmaxf(rmax[j], __shfl_xor(rmax[j], 2, 32));
            rmax[j] = fmaxf(rmax[j], __shfl_xor(rmax[j], 4, 32));
            rmax[j] = fmaxf(rmax[j], __shfl_xor(rmax[j], 8, 32));
        }

        // ---- online softmax update ----
        float alpha[8], psum[8];
#pragma unroll
        for (int j = 0; j < 8; ++j) {
            const float nm = fmaxf(mrow[j], rmax[j]);
            alpha[j] = __expf(mrow[j] - nm);
            mrow[j] = nm;
            psum[j] = 0.f;
        }
#pragma unroll
        for (int nt = 0; nt < 2; ++nt)
#pragma unroll
            for (int j = 0; j < 8; ++j) {
                const float p = __expf(sacc[nt][j] - mrow[j]);
                sacc[nt][j] = p;
                psum[j] += p;
            }
#pragma unroll
        for (int j = 0; j < 8; ++j) {
            psum[j] += __shfl_xor(psum[j], 1, 32);
            psum[j] += __shfl_xor(psum[j], 2, 32);
            psum[j] += __shfl_xor(psum[j], 4, 32);
            psum[j] += __shfl_xor(psum[j], 8, 32);
            lrow[j] = lrow[j] * alpha[j] + psum[j];
        }
#pragma unroll
        for (int i = 0; i < 4; ++i)
#pragma unroll
            for (int j = 0; j < 8; ++j) oacc[i][j] *= alpha[j];

        // ---- P: C-layout -> A-fragment layout via LDS bounce ----
#pragma unroll
        for (int nt = 0; nt < 2; ++nt)
#pragma unroll
            for (int j = 0; j < 8; ++j)
                pls[half * 8 + j][nt * 16 + lane_lo] = tobf(sacc[nt][j]);
        asm volatile("s_wait_dscnt 0x0" ::: "memory");
        BF16x16 ap;
#pragma unroll
        for (int v = 0; v < 8; ++v) {
            const int kb = (v >> 2) * 16 + half * 8 + (v & 3) * 2;
            ap.u[v] = *(const unsigned*)&pls[lane_lo][kb];
        }
        asm volatile("" ::: "memory");

        // ---- O += P(16x32) * V(32x64) ----
#pragma unroll
        for (int i = 0; i < 4; ++i) {
            BF16x16 bv;
#pragma unroll
            for (int v = 0; v < 8; ++v) {
                const int k0 = (v >> 2) * 16 + half * 8 + (v & 3) * 2;
                const unsigned short lo =
                    *(const unsigned short*)(Vb + (size_t)(t0 + k0) * Cq + i * 16 + lane_lo);
                const unsigned short hi =
                    *(const unsigned short*)(Vb + (size_t)(t0 + k0 + 1) * Cq + i * 16 + lane_lo);
                bv.u[v] = (unsigned)lo | ((unsigned)hi << 16);
            }
            oacc[i] = __builtin_amdgcn_wmma_f32_16x16x32_bf16(
                false, ap.v, false, bv.v, (short)0, oacc[i], false, false);
        }
    }

    // ---- normalize + store (heads concatenated row-major) ----
#pragma unroll
    for (int i = 0; i < 4; ++i)
#pragma unroll
        for (int j = 0; j < 8; ++j) {
            const int row = q0 + half * 8 + j;
            const float ov = oacc[i][j] / lrow[j];
            O[((size_t)b * Sq + row) * Cq + (size_t)h * HDq + i * 16 + lane_lo] = tobf(ov);
        }
}

// ---------------------------------------------------------------------------
// launcher
// ---------------------------------------------------------------------------
extern "C" void kernel_launch(void* const* d_in, const int* in_sizes, int n_in,
                              void* d_out, int out_size, void* d_ws, size_t ws_size,
                              hipStream_t stream)
{
    (void)in_sizes; (void)n_in; (void)out_size; (void)ws_size;

    const int*   idx   = (const int*)d_in[0];
    const float* tok   = (const float*)d_in[1];
    const float* pos   = (const float*)d_in[2];
    const float* Wq    = (const float*)d_in[3];
    const float* Wk    = (const float*)d_in[4];
    const float* Wv    = (const float*)d_in[5];
    const float* Wproj = (const float*)d_in[6];
    const float* bproj = (const float*)d_in[7];
    const float* W1    = (const float*)d_in[8];
    const float* b1    = (const float*)d_in[9];
    const float* W2    = (const float*)d_in[10];
    const float* b2    = (const float*)d_in[11];
    const float* Wlm   = (const float*)d_in[12];
    const float* blm   = (const float*)d_in[13];

    char* ws = (char*)d_ws;
    size_t off = 0;
    auto alloc = [&](size_t bytes) {
        void* p = ws + off;
        off = (off + bytes + 255) & ~(size_t)255;
        return p;
    };

    const size_t Mr = (size_t)Bq * Sq;               // 4096 rows
    bf16_t* xbf = (bf16_t*)alloc(Mr * Cq * 2);
    bf16_t* Qbf = (bf16_t*)alloc(Mr * Cq * 2);
    bf16_t* Kbf = (bf16_t*)alloc(Mr * Cq * 2);
    bf16_t* Vbf = (bf16_t*)alloc(Mr * Cq * 2);
    bf16_t* Obf = (bf16_t*)alloc(Mr * Cq * 2);
    bf16_t* Ybf = (bf16_t*)alloc(Mr * Cq * 2);
    bf16_t* Hbf = (bf16_t*)alloc(Mr * 4 * Cq * 2);
    bf16_t* Zbf = (bf16_t*)alloc(Mr * Cq * 2);
    bf16_t* wq  = (bf16_t*)alloc((size_t)Cq * Cq * 2);
    bf16_t* wk  = (bf16_t*)alloc((size_t)Cq * Cq * 2);
    bf16_t* wv  = (bf16_t*)alloc((size_t)Cq * Cq * 2);
    bf16_t* wpj = (bf16_t*)alloc((size_t)Cq * Cq * 2);
    bf16_t* w1  = (bf16_t*)alloc((size_t)Cq * 4 * Cq * 2);
    bf16_t* w2  = (bf16_t*)alloc((size_t)4 * Cq * Cq * 2);
    bf16_t* wlm = (bf16_t*)alloc((size_t)Cq * Vq * 2);

    auto nblocks = [](size_t n) { return (unsigned)((n + 255) / 256); };

    // embedding + weight conversion
    embed_kernel<<<nblocks(Mr * Cq), 256, 0, stream>>>(idx, tok, pos, xbf);
    repack_qkv_kernel<<<nblocks((size_t)Hq * Cq * HDq), 256, 0, stream>>>(Wq, wq);
    repack_qkv_kernel<<<nblocks((size_t)Hq * Cq * HDq), 256, 0, stream>>>(Wk, wk);
    repack_qkv_kernel<<<nblocks((size_t)Hq * Cq * HDq), 256, 0, stream>>>(Wv, wv);
    cvt_kernel<<<nblocks((size_t)Cq * Cq),     256, 0, stream>>>(Wproj, wpj, (long)Cq * Cq);
    cvt_kernel<<<nblocks((size_t)Cq * 4 * Cq), 256, 0, stream>>>(W1,    w1,  (long)Cq * 4 * Cq);
    cvt_kernel<<<nblocks((size_t)4 * Cq * Cq), 256, 0, stream>>>(W2,    w2,  (long)4 * Cq * Cq);
    cvt_kernel<<<nblocks((size_t)Cq * Vq),     256, 0, stream>>>(Wlm,   wlm, (long)Cq * Vq);

    const dim3 gC(Cq / 128, (unsigned)(Mr / 128));          // N=1024
    const dim3 gF(4 * Cq / 128, (unsigned)(Mr / 128));      // N=4096
    const dim3 gV(Vq / 128, (unsigned)(Mr / 128));          // N=32000

    // QKV projections
    gemm_kernel<0, true><<<gC, 256, 0, stream>>>(xbf, wq, nullptr, Qbf, (int)Mr, Cq, Cq);
    gemm_kernel<0, true><<<gC, 256, 0, stream>>>(xbf, wk, nullptr, Kbf, (int)Mr, Cq, Cq);
    gemm_kernel<0, true><<<gC, 256, 0, stream>>>(xbf, wv, nullptr, Vbf, (int)Mr, Cq, Cq);

    // causal multi-head attention
    attn_kernel<<<Bq * Hq * (Sq / 16), 32, 0, stream>>>(Qbf, Kbf, Vbf, Obf);

    // output projection, FFN, LM head
    gemm_kernel<1, true ><<<gC, 256, 0, stream>>>(Obf, wpj, bproj, Ybf, (int)Mr, Cq, Cq);
    gemm_kernel<2, true ><<<gF, 256, 0, stream>>>(Ybf, w1,  b1,   Hbf, (int)Mr, 4 * Cq, Cq);
    gemm_kernel<1, true ><<<gC, 256, 0, stream>>>(Hbf, w2,  b2,   Zbf, (int)Mr, Cq, 4 * Cq);
    gemm_kernel<1, false><<<gV, 256, 0, stream>>>(Zbf, wlm, blm,  d_out, (int)Mr, Vq, Cq);
}
